// LSTMModel_53678501265695
// MI455X (gfx1250) — matrix operand order repaired and checked by reading, not compiled, not run on hardware
//
#include <hip/hip_runtime.h>
#include <math.h>

typedef _Float16 half_t;
typedef __attribute__((ext_vector_type(8)))  _Float16 v8h;
typedef __attribute__((ext_vector_type(16))) _Float16 v16h;
typedef __attribute__((ext_vector_type(8)))  float    v8f;

#define SEQ 64
#define BN  32
#define EDIM 512
#define HDIM 1024
#define VOC 32000
#define G4  (4*HDIM)
#define MB  (SEQ*BN)   // 2048

// ---------------- fp32 -> fp16 convert ----------------
__global__ void cvt_f32_f16(const float* __restrict__ src, half_t* __restrict__ dst, size_t n) {
    size_t i = (size_t)blockIdx.x * blockDim.x + threadIdx.x;
    if (i < n) dst[i] = (half_t)src[i];
}

// ---------------- embedding gather (fp32 table -> f16 rows) ----------------
__global__ void embed_gather(const float* __restrict__ emb, const int* __restrict__ x,
                             half_t* __restrict__ out) {
    int row = blockIdx.x;                 // row = s*BN + b
    int idx = x[row];
    const float* src = emb + (size_t)idx * EDIM;
    half_t* dst = out + (size_t)row * EDIM;
    for (int c = threadIdx.x; c < EDIM; c += blockDim.x) dst[c] = (half_t)src[c];
}

// ---------------- big WMMA GEMM: Cout[M,N] = A[M,K] * Bw[N,K]^T (+bias1)(+bias2)
// One wave per block computes a 64x64 output tile (4 M-subtiles x 4 N-subtiles):
// each B fragment is reused across 4 WMMAs -> 4x less L2 traffic on the weight
// matrix than a 16-row tile.  All fragment loads are contiguous b128 chunks.
__global__ void __launch_bounds__(32)
wmma_gemm64x64(const half_t* __restrict__ A, const half_t* __restrict__ Bw,
               const float* __restrict__ bias1, const float* __restrict__ bias2,
               float* __restrict__ Cout, int M, int N, int K) {
    const int lane = threadIdx.x;
    const int l16  = lane & 15;
    const int hi   = lane >> 4;
    const int m0   = blockIdx.y * 64;
    const int n0   = blockIdx.x * 64;

    v8f acc[4][4] = {};

    const half_t* arow[4];
    const half_t* brow[4];
#pragma unroll
    for (int mt = 0; mt < 4; ++mt) arow[mt] = A  + (size_t)(m0 + mt * 16 + l16) * K;
#pragma unroll
    for (int t = 0; t < 4; ++t)    brow[t]  = Bw + (size_t)(n0 + t  * 16 + l16) * K;

    for (int k0 = 0; k0 < K; k0 += 32) {
        // A fragment: lanes 0-15 hold K=0..15, lanes 16-31 hold K=16..31
        // (same k order as B) -> two contiguous 16B chunks per lane.
        union { v16h v; v8h h[2]; } a[4];
#pragma unroll
        for (int mt = 0; mt < 4; ++mt) {
            a[mt].h[0] = *(const v8h*)(arow[mt] + k0 + 16 * hi);
            a[mt].h[1] = *(const v8h*)(arow[mt] + k0 + 16 * hi + 8);
        }
#pragma unroll
        for (int t = 0; t < 4; ++t) {
            // B fragment: lanes 0-15 hold K=0..15, lanes 16-31 K=16..31 -> one 32B chunk.
            v16h b = *(const v16h*)(brow[t] + k0 + 16 * hi);
            if (k0 + 32 < K) __builtin_prefetch(brow[t] + k0 + 32 + 16 * hi, 0, 1);
#pragma unroll
            for (int mt = 0; mt < 4; ++mt)
                acc[mt][t] = __builtin_amdgcn_wmma_f32_16x16x32_f16(
                    false, a[mt].v, false, b, (short)0, acc[mt][t], false, false);
        }
    }

    // C/D layout: VGPR r -> row r (lanes 0-15) / row r+8 (lanes 16-31); col = l16
#pragma unroll
    for (int t = 0; t < 4; ++t) {
        int n = n0 + t * 16 + l16;
        float bsum = 0.f;
        if (bias1) bsum += bias1[n];
        if (bias2) bsum += bias2[n];
#pragma unroll
        for (int mt = 0; mt < 4; ++mt) {
#pragma unroll
            for (int r = 0; r < 8; ++r) {
                int m = m0 + mt * 16 + r + 8 * hi;
                Cout[(size_t)m * N + n] = acc[mt][t][r] + bsum;
            }
        }
    }
}

// ---------------- sigmoid helper ----------------
__device__ __forceinline__ float sigf(float x) { return 1.0f / (1.0f + expf(-x)); }

// ---------------- device-wide barrier (monotonic counter) ----------------
__device__ __forceinline__ void grid_barrier(unsigned* ctr, unsigned target) {
    __threadfence();                                   // release h stores
    if (threadIdx.x == 0) {
        atomicAdd(ctr, 1u);
        while (__hip_atomic_load(ctr, __ATOMIC_RELAXED, __HIP_MEMORY_SCOPE_AGENT) < target)
            __builtin_amdgcn_s_sleep(2);
    }
    __threadfence();                                   // acquire before reading h
}

// ---------------- persistent LSTM layer ----------------
// Grid = 64 blocks x 1 wave.  Wave w owns output columns j0 = 16*w.
// Its 64 W_hh rows (4 gates x 16 cols x K=1024, 128KB f16) live in LDS for all
// 64 timesteps (MI455X: 320KB LDS per WGP).  Cell state c stays in registers.
// h is double-buffered through y[s]; one device-wide barrier per step.
__global__ void __launch_bounds__(32)
lstm_layer_persistent(const half_t* __restrict__ Whh,   // [4H, H] f16
                      const float*  __restrict__ Xg,    // [S, B, 4H] precomputed input gates
                      const float*  __restrict__ c0,    // [B, H] initial c (layer slice)
                      const half_t* __restrict__ h0h,   // [B, H] initial h, f16
                      float* __restrict__ cOut,         // [B, H] final c
                      float* __restrict__ hOut,         // [B, H] final h
                      half_t* __restrict__ y,           // [S, B, H] f16 layer output
                      unsigned* __restrict__ ctr) {
    extern __shared__ char smem[];
    half_t* Wlds = (half_t*)smem;          // [4*16][HDIM]
    const int lane = threadIdx.x;
    const int l16  = lane & 15;
    const int hi   = lane >> 4;
    const int j0   = blockIdx.x * 16;      // output column range [j0, j0+16)

    // ---- one-time: stage the 64 needed W_hh rows into LDS ----
    for (int rr = 0; rr < 64; ++rr) {
        int g = rr >> 4, r = rr & 15;
        const half_t* src = Whh + (size_t)(g * HDIM + j0 + r) * HDIM;
        half_t* dst = Wlds + (size_t)rr * HDIM;
        for (int k = lane * 8; k < HDIM; k += 32 * 8)
            *(v8h*)(dst + k) = *(const v8h*)(src + k);
    }

    // ---- cell state in registers (C-fragment layout) ----
    float cReg[2][8];
#pragma unroll
    for (int mt = 0; mt < 2; ++mt)
#pragma unroll
        for (int r = 0; r < 8; ++r) {
            int m = mt * 16 + r + 8 * hi;
            cReg[mt][r] = c0[(size_t)m * HDIM + j0 + l16];
        }

    for (int s = 0; s < SEQ; ++s) {
        const half_t* hPrev = (s == 0) ? h0h : (y + (size_t)(s - 1) * BN * HDIM);

        v8f acc[4][2] = {};    // [gate][m-subtile]
        for (int k0 = 0; k0 < HDIM; k0 += 32) {
            union { v16h v; v8h h[2]; } a[2];
#pragma unroll
            for (int mt = 0; mt < 2; ++mt) {
                const half_t* ar = hPrev + (size_t)(mt * 16 + l16) * HDIM;
                a[mt].h[0] = *(const v8h*)(ar + k0 + 16 * hi);
                a[mt].h[1] = *(const v8h*)(ar + k0 + 16 * hi + 8);
            }
#pragma unroll
            for (int g = 0; g < 4; ++g) {
                v16h b = *(const v16h*)(Wlds + (size_t)(g * 16 + l16) * HDIM + k0 + 16 * hi);
                acc[g][0] = __builtin_amdgcn_wmma_f32_16x16x32_f16(
                    false, a[0].v, false, b, (short)0, acc[g][0], false, false);
                acc[g][1] = __builtin_amdgcn_wmma_f32_16x16x32_f16(
                    false, a[1].v, false, b, (short)0, acc[g][1], false, false);
            }
        }

        // ---- fused cell: gates + nonlinearity + state update ----
#pragma unroll
        for (int mt = 0; mt < 2; ++mt) {
#pragma unroll
            for (int r = 0; r < 8; ++r) {
                int m = mt * 16 + r + 8 * hi;                 // m == batch index
                const float* xrow = Xg + ((size_t)s * BN + m) * G4 + j0 + l16;
                float gi = acc[0][mt][r] + xrow[0];
                float gf = acc[1][mt][r] + xrow[HDIM];
                float gg = acc[2][mt][r] + xrow[2 * HDIM];
                float go = acc[3][mt][r] + xrow[3 * HDIM];
                float cn = sigf(gf) * cReg[mt][r] + sigf(gi) * tanhf(gg);
                float hn = sigf(go) * tanhf(cn);
                cReg[mt][r] = cn;
                y[((size_t)s * BN + m) * HDIM + j0 + l16] = (half_t)hn;
                if (s == SEQ - 1) {
                    hOut[(size_t)m * HDIM + j0 + l16] = hn;
                    cOut[(size_t)m * HDIM + j0 + l16] = cn;
                }
            }
        }

        if (s != SEQ - 1)
            grid_barrier(ctr, (unsigned)((s + 1) * gridDim.x));
    }
}

// ---------------- in-place log_softmax over rows of length VOC ----------------
__global__ void log_softmax_rows(float* __restrict__ logits) {
    __shared__ float red[256];
    int tid = threadIdx.x;
    float* row = logits + (size_t)blockIdx.x * VOC;

    float m = -3.4e38f;
    for (int j = tid; j < VOC; j += 256) m = fmaxf(m, row[j]);
    red[tid] = m; __syncthreads();
    for (int s = 128; s > 0; s >>= 1) { if (tid < s) red[tid] = fmaxf(red[tid], red[tid + s]); __syncthreads(); }
    m = red[0]; __syncthreads();

    float sum = 0.f;
    for (int j = tid; j < VOC; j += 256) sum += expf(row[j] - m);
    red[tid] = sum; __syncthreads();
    for (int s = 128; s > 0; s >>= 1) { if (tid < s) red[tid] += red[tid + s]; __syncthreads(); }
    float lse = m + logf(red[0]);

    for (int j = tid; j < VOC; j += 256) row[j] = row[j] - lse;
}

// ---------------- host-side orchestration ----------------
extern "C" void kernel_launch(void* const* d_in, const int* in_sizes, int n_in,
                              void* d_out, int out_size, void* d_ws, size_t ws_size,
                              hipStream_t stream) {
    (void)in_sizes; (void)n_in; (void)out_size; (void)ws_size;

    const int*   x     = (const int*)  d_in[0];
    const float* h0    = (const float*)d_in[1];
    const float* c0    = (const float*)d_in[2];
    const float* emb   = (const float*)d_in[3];
    const float* W_ih0 = (const float*)d_in[4];
    const float* W_hh0 = (const float*)d_in[5];
    const float* b_ih0 = (const float*)d_in[6];
    const float* b_hh0 = (const float*)d_in[7];
    const float* W_ih1 = (const float*)d_in[8];
    const float* W_hh1 = (const float*)d_in[9];
    const float* b_ih1 = (const float*)d_in[10];
    const float* b_hh1 = (const float*)d_in[11];
    const float* fc_W  = (const float*)d_in[12];
    const float* fc_b  = (const float*)d_in[13];

    char* ws = (char*)d_ws;
    size_t off = 0;
    auto take = [&](size_t bytes) -> char* {
        char* p = ws + off;
        off += (bytes + 255) & ~(size_t)255;
        return p;
    };

    half_t*   Wih0h = (half_t*)take((size_t)G4 * EDIM * 2);
    half_t*   Whh0h = (half_t*)take((size_t)G4 * HDIM * 2);
    half_t*   Wih1h = (half_t*)take((size_t)G4 * HDIM * 2);
    half_t*   Whh1h = (half_t*)take((size_t)G4 * HDIM * 2);
    half_t*   fcWh  = (half_t*)take((size_t)VOC * HDIM * 2);
    half_t*   eh    = (half_t*)take((size_t)MB * EDIM * 2);
    float*    Xg0   = (float*) take((size_t)MB * G4 * 4);
    float*    Xg1   = (float*) take((size_t)MB * G4 * 4);
    half_t*   y0h   = (half_t*)take((size_t)MB * HDIM * 2);
    half_t*   y1h   = (half_t*)take((size_t)MB * HDIM * 2);
    half_t*   h0h   = (half_t*)take((size_t)2 * BN * HDIM * 2);
    unsigned* ctrs  = (unsigned*)take(256);   // two barrier counters

    float* out = (float*)d_out;
    const size_t np = (size_t)MB * VOC;
    float* hN = out + np;                       // [2, B, H]
    float* cN = out + np + (size_t)2 * BN * HDIM;

    const size_t LDS_BYTES = (size_t)4 * 16 * HDIM * 2;   // 128 KB
    hipFuncSetAttribute(reinterpret_cast<const void*>(lstm_layer_persistent),
                        hipFuncAttributeMaxDynamicSharedMemorySize, (int)LDS_BYTES);

    auto cvt = [&](const float* s, half_t* d, size_t n) {
        cvt_f32_f16<<<dim3((unsigned)((n + 255) / 256)), dim3(256), 0, stream>>>(s, d, n);
    };

    // 0) barrier counters must start at zero each call
    hipMemsetAsync(ctrs, 0, 256, stream);

    // 1) weight / state conversions to f16
    cvt(W_ih0, Wih0h, (size_t)G4 * EDIM);
    cvt(W_hh0, Whh0h, (size_t)G4 * HDIM);
    cvt(W_ih1, Wih1h, (size_t)G4 * HDIM);
    cvt(W_hh1, Whh1h, (size_t)G4 * HDIM);
    cvt(fc_W,  fcWh,  (size_t)VOC * HDIM);
    cvt(h0,    h0h,   (size_t)2 * BN * HDIM);

    // 2) embedding gather
    embed_gather<<<dim3(MB), dim3(256), 0, stream>>>(emb, x, eh);

    // 3) layer-0 input gates: Xg0 = e @ W_ih0^T + b_ih0 + b_hh0
    wmma_gemm64x64<<<dim3(G4 / 64, MB / 64), dim3(32), 0, stream>>>(
        eh, Wih0h, b_ih0, b_hh0, Xg0, MB, G4, EDIM);

    // 4) layer-0 recurrence: single persistent kernel, W_hh LDS-resident
    lstm_layer_persistent<<<dim3(HDIM / 16), dim3(32), LDS_BYTES, stream>>>(
        Whh0h, Xg0, c0, h0h, cN, hN, y0h, ctrs);

    // 5) layer-1 input gates: Xg1 = y0 @ W_ih1^T + b_ih1 + b_hh1
    wmma_gemm64x64<<<dim3(G4 / 64, MB / 64), dim3(32), 0, stream>>>(
        y0h, Wih1h, b_ih1, b_hh1, Xg1, MB, G4, HDIM);

    // 6) layer-1 recurrence
    lstm_layer_persistent<<<dim3(HDIM / 16), dim3(32), LDS_BYTES, stream>>>(
        Whh1h, Xg1, c0 + (size_t)BN * HDIM, h0h + (size_t)BN * HDIM,
        cN + (size_t)BN * HDIM, hN + (size_t)BN * HDIM, y1h, ctrs + 64);

    // 7) FC projection straight into d_out: logits = y1 @ fc_W^T + fc_b
    wmma_gemm64x64<<<dim3(VOC / 64, MB / 64), dim3(32), 0, stream>>>(
        y1h, fcWh, fc_b, nullptr, out, MB, VOC, HDIM);

    // 8) in-place log_softmax over each of the 2048 rows
    log_softmax_rows<<<dim3(MB), dim3(256), 0, stream>>>(out);
}
